// MultiHeadAttention_53180285059815
// MI455X (gfx1250) — matrix operand order repaired and checked by reading, not compiled, not run on hardware
//
#include <hip/hip_runtime.h>
#include <hip/hip_bf16.h>

// ---- sizes ----
#define BB   4
#define SS   2048
#define CC   1024
#define HH   16
#define DD   64

typedef __attribute__((ext_vector_type(16))) _Float16 v16h;
typedef __attribute__((ext_vector_type(8)))  _Float16 v8h;
typedef __attribute__((ext_vector_type(8)))  float    v8f;

__device__ __forceinline__ v8h ld8(const _Float16* p) { return *(const v8h*)p; }

__device__ __forceinline__ v16h frag16(v8h a, v8h b) {
    v16h r;
#pragma unroll
    for (int i = 0; i < 8; ++i) { r[i] = a[i]; r[8 + i] = b[i]; }
    return r;
}

__device__ __forceinline__ v8f wmma_f16(v16h a, v16h b, v8f c) {
    // D = A(16x32 f16) x B(32x16 f16) + C(16x16 f32)
    return __builtin_amdgcn_wmma_f32_16x16x32_f16(false, a, false, b, (short)0, c,
                                                  false, false);
}

__device__ __forceinline__ v8f zero8() {
    v8f z = {0.f, 0.f, 0.f, 0.f, 0.f, 0.f, 0.f, 0.f};
    return z;
}

// ---- CDNA5 async global->LDS (ASYNCcnt) helpers ----
// ISA 10.2: for an LDS generic address, addr[31:0] is the LDS byte address,
// so the low 32 bits of a __shared__ pointer form the VDST operand.
__device__ __forceinline__ void async_ld_b128(void* lds_ptr, const void* gaddr) {
    unsigned lds = (unsigned)(size_t)lds_ptr;
    asm volatile("global_load_async_to_lds_b128 %0, %1, off"
                 :: "v"(lds), "v"(gaddr) : "memory");
}
__device__ __forceinline__ void wait_async0() {
    asm volatile("s_wait_asynccnt 0" ::: "memory");
}

// ---------- convert fp32 -> fp16, 8 elems/thread ----------
__global__ void __launch_bounds__(256)
cvt_f32_to_f16(const float* __restrict__ in, _Float16* __restrict__ out, size_t n) {
    size_t i = ((size_t)blockIdx.x * blockDim.x + threadIdx.x) * 8;
    if (i + 8 <= n) {
        float4 a = *(const float4*)(in + i);
        float4 b = *(const float4*)(in + i + 4);
        v8h o;
        o[0] = (_Float16)a.x; o[1] = (_Float16)a.y;
        o[2] = (_Float16)a.z; o[3] = (_Float16)a.w;
        o[4] = (_Float16)b.x; o[5] = (_Float16)b.y;
        o[6] = (_Float16)b.z; o[7] = (_Float16)b.w;
        *(v8h*)(out + i) = o;
    }
}

// ---------- transpose W[rows][cols] f32 -> Wt[cols][rows] f16 ----------
__global__ void __launch_bounds__(256)
transpose_f32_to_f16(const float* __restrict__ W, _Float16* __restrict__ Wt,
                     int rows, int cols) {
    __shared__ float tile[32][33];
    int c0 = blockIdx.x * 32;   // source col base (k)
    int r0 = blockIdx.y * 32;   // source row base (n)
    int tx = threadIdx.x, ty = threadIdx.y;  // block (32, 8)
#pragma unroll
    for (int i = 0; i < 32; i += 8)
        tile[ty + i][tx] = W[(size_t)(r0 + ty + i) * cols + (c0 + tx)];
    __syncthreads();
#pragma unroll
    for (int i = 0; i < 32; i += 8)
        Wt[(size_t)(c0 + ty + i) * rows + (r0 + tx)] = (_Float16)tile[tx][ty + i];
}

// ---------- QKV projection GEMM ----------
// B tile (32k x 64n f16 = 4KB) is shared by all 8 waves: stage it via the
// async DMA into LDS, double-buffered, one barrier per k-step.
__global__ void __launch_bounds__(256)
gemm_qkv(const _Float16* __restrict__ xh, const _Float16* __restrict__ wt,
         const float* __restrict__ bias,
         _Float16* __restrict__ Qh, _Float16* __restrict__ Kh,
         _Float16* __restrict__ Vh) {
    __shared__ _Float16 bsm[2][32 * 64];
    int tid = threadIdx.x;
    int lane = tid & 31, wid = tid >> 5;
    int lo = lane & 15, hi = lane >> 4;
    int wg = blockIdx.x * 8 + wid;          // 24576 waves
    int m0 = (wg & 511) * 16;               // 512 row tiles
    int n0 = (wg >> 9) * 64;                // 48 col groups of 64
    int trow = tid >> 3;                    // async slice: k row 0..31
    int tcol = (tid & 7) * 8;               // async slice: 8 halves (16B)

    v8f acc[4];
#pragma unroll
    for (int nt = 0; nt < 4; ++nt) acc[nt] = zero8();

    // prologue: DMA B tile for kc=0
    async_ld_b128(&bsm[0][trow * 64 + tcol],
                  wt + (size_t)trow * 3072 + n0 + tcol);

    const _Float16* arow = xh + (size_t)(m0 + lo) * 1024;
    for (int kc = 0; kc < 1024; kc += 32) {
        int buf = (kc >> 5) & 1;
        wait_async0();
        __syncthreads();
        if (kc + 32 < 1024)
            async_ld_b128(&bsm[buf ^ 1][trow * 64 + tcol],
                          wt + (size_t)(kc + 32 + trow) * 3072 + n0 + tcol);

        const _Float16* ap = arow + kc + 8 * hi;         // A: lane=M, elems=K
        v16h a = frag16(ld8(ap), ld8(ap + 16));
        const _Float16* bp = &bsm[buf][lo];  // B: lane=N, elems=K
#pragma unroll
        for (int nt = 0; nt < 4; ++nt) {
            v16h b;
#pragma unroll
            for (int j = 0; j < 16; ++j)
                b[j] = bp[((j & 7) + 8 * hi + 16 * (j >> 3)) * 64 + nt * 16];
            acc[nt] = wmma_f16(a, b, acc[nt]);
        }
    }
#pragma unroll
    for (int nt = 0; nt < 4; ++nt) {
        int n = n0 + nt * 16 + lo;          // D layout: lane = N
        float bv = bias[n];
        int sec = n >> 10, cc = n & 1023;
        int h = cc >> 6, d = cc & 63;
        _Float16* dst = (sec == 0) ? Qh : ((sec == 1) ? Kh : Vh);
        float scl = (sec == 0) ? 0.125f : 1.0f;   // HEAD_DIM^-0.5 folded into Q
#pragma unroll
        for (int r = 0; r < 8; ++r) {        // D layout: VGPR r -> row r+8*hi
            int m = m0 + r + 8 * hi;
            int b_ = m >> 11, s = m & 2047;
            dst[(((size_t)(b_ * HH + h)) * SS + s) * DD + d] =
                (_Float16)((acc[nt][r] + bv) * scl);
        }
    }
}

// ---------- flash attention ----------
// One wave = 16 query rows. S^T = K * Q^T so the softmax probabilities land
// directly in the A-operand layout for P*V (lane = s, regs = t).
// K/V 32x64 tiles staged per-workgroup via async DMA into LDS, double-buffered.
__global__ void __launch_bounds__(256)
attn_kernel(const _Float16* __restrict__ Q, const _Float16* __restrict__ K,
            const _Float16* __restrict__ V, _Float16* __restrict__ Oh) {
    __shared__ _Float16 smem[2][2][32 * 64];   // [buf][0=K,1=V]
    int tid = threadIdx.x;
    int lane = tid & 31, wid = tid >> 5;
    int lo = lane & 15, hi = lane >> 4;
    int bh = blockIdx.x >> 4;               // 64 (b,h) pairs
    int sblk = blockIdx.x & 15;
    int s0 = sblk * 128 + wid * 16;
    int trow = tid >> 3;                    // async slice: t row 0..31
    int tcol = (tid & 7) * 8;               // async slice: 8 halves (16B)

    const _Float16* Qp = Q + (size_t)bh * SS * DD;
    const _Float16* Kp = K + (size_t)bh * SS * DD;
    const _Float16* Vp = V + (size_t)bh * SS * DD;

    // Q^T as B-operand: lane = s (N dim), elems = d (K dim). Gathered once.
    v16h qb[2];
#pragma unroll
    for (int c = 0; c < 2; ++c) {
#pragma unroll
        for (int j = 0; j < 16; ++j)
            qb[c][j] = Qp[(size_t)(s0 + lo) * DD + c * 32 + (j & 7) + 8 * hi + 16 * (j >> 3)];
    }

    float mrow = -1e30f, lrow = 0.f;
    v8f o[4];
#pragma unroll
    for (int dt = 0; dt < 4; ++dt) o[dt] = zero8();

    // prologue: DMA K/V tiles for tb=0
    async_ld_b128(&smem[0][0][trow * 64 + tcol], Kp + (size_t)trow * DD + tcol);
    async_ld_b128(&smem[0][1][trow * 64 + tcol], Vp + (size_t)trow * DD + tcol);

    for (int tb = 0; tb < SS; tb += 32) {
        int buf = (tb >> 5) & 1;
        wait_async0();
        __syncthreads();
        if (tb + 32 < SS) {
            int nb = buf ^ 1, t2 = tb + 32;
            async_ld_b128(&smem[nb][0][trow * 64 + tcol],
                          Kp + (size_t)(t2 + trow) * DD + tcol);
            async_ld_b128(&smem[nb][1][trow * 64 + tcol],
                          Vp + (size_t)(t2 + trow) * DD + tcol);
        }
        const _Float16* ksm = smem[buf][0];
        const _Float16* vsm = smem[buf][1];

        // S^T tiles: tau covers t = tb+16*tau .. +15
        v8f st[2];
#pragma unroll
        for (int tau = 0; tau < 2; ++tau) {
            st[tau] = zero8();
#pragma unroll
            for (int c = 0; c < 2; ++c) {
                const _Float16* kp = ksm + (16 * tau + lo) * 64 + c * 32 + 8 * hi;
                v16h ka = frag16(ld8(kp), ld8(kp + 16));   // A: lane=t, elems=d
                st[tau] = wmma_f16(ka, qb[c], st[tau]);
            }
        }
        // online softmax over this 32-wide t slab; row = s = lo
        float mloc = st[0][0];
#pragma unroll
        for (int r = 0; r < 8; ++r) {
            mloc = fmaxf(mloc, st[0][r]);
            mloc = fmaxf(mloc, st[1][r]);
        }
        mloc = fmaxf(mloc, __shfl_xor(mloc, 16, 32));
        float mnew = fmaxf(mrow, mloc);
        float scale = __expf(mrow - mnew);
        float psum = 0.f;
        v16h pa;   // P in A-operand layout: lane=s, elems=t (matches D of S^T)
#pragma unroll
        for (int r = 0; r < 8; ++r) {
            float p0 = __expf(st[0][r] - mnew);
            float p1 = __expf(st[1][r] - mnew);
            psum += p0 + p1;
            pa[r] = (_Float16)p0;
            pa[8 + r] = (_Float16)p1;
        }
        psum += __shfl_xor(psum, 16, 32);
        lrow = lrow * scale + psum;
        mrow = mnew;
        // rescale accumulators: O rows are indexed by VGPR (r + 8*hi)
#pragma unroll
        for (int r = 0; r < 8; ++r) {
            float sc = __shfl(scale, r + 8 * hi, 32);
            o[0][r] *= sc; o[1][r] *= sc; o[2][r] *= sc; o[3][r] *= sc;
        }
        // O += P * V ; V as B-operand: lane = d, elems = t
#pragma unroll
        for (int dt = 0; dt < 4; ++dt) {
            const _Float16* vp = vsm + dt * 16 + lo;
            v16h vb;
#pragma unroll
            for (int j = 0; j < 16; ++j)
                vb[j] = vp[((j & 7) + 8 * hi + 16 * (j >> 3)) * 64];
            o[dt] = wmma_f16(pa, vb, o[dt]);
        }
    }
    // normalize and store O as [B][S][C] f16 (heads concatenated)
    int b = bh >> 4, h = bh & 15;
#pragma unroll
    for (int r = 0; r < 8; ++r) {
        float inv = 1.0f / __shfl(lrow, r + 8 * hi, 32);
        int s = s0 + r + 8 * hi;
        size_t rowbase = ((size_t)b * SS + s) * CC + h * DD + lo;
#pragma unroll
        for (int dt = 0; dt < 4; ++dt)
            Oh[rowbase + dt * 16] = (_Float16)(o[dt][r] * inv);
    }
}

// ---------- output projection GEMM ----------
__global__ void __launch_bounds__(256)
gemm_out(const _Float16* __restrict__ Oh, const _Float16* __restrict__ wt,
         const float* __restrict__ bias, float* __restrict__ out) {
    __shared__ _Float16 bsm[2][32 * 64];
    int tid = threadIdx.x;
    int lane = tid & 31, wid = tid >> 5;
    int lo = lane & 15, hi = lane >> 4;
    int wg = blockIdx.x * 8 + wid;          // 8192 waves
    int m0 = (wg & 511) * 16;
    int n0 = (wg >> 9) * 64;                // 16 col groups
    int trow = tid >> 3;
    int tcol = (tid & 7) * 8;

    v8f acc[4];
#pragma unroll
    for (int nt = 0; nt < 4; ++nt) acc[nt] = zero8();

    async_ld_b128(&bsm[0][trow * 64 + tcol],
                  wt + (size_t)trow * 1024 + n0 + tcol);

    const _Float16* arow = Oh + (size_t)(m0 + lo) * 1024;
    for (int kc = 0; kc < 1024; kc += 32) {
        int buf = (kc >> 5) & 1;
        wait_async0();
        __syncthreads();
        if (kc + 32 < 1024)
            async_ld_b128(&bsm[buf ^ 1][trow * 64 + tcol],
                          wt + (size_t)(kc + 32 + trow) * 1024 + n0 + tcol);

        const _Float16* ap = arow + kc + 8 * hi;
        v16h a = frag16(ld8(ap), ld8(ap + 16));
        const _Float16* bp = &bsm[buf][lo];
#pragma unroll
        for (int nt = 0; nt < 4; ++nt) {
            v16h b;
#pragma unroll
            for (int j = 0; j < 16; ++j)
                b[j] = bp[((j & 7) + 8 * hi + 16 * (j >> 3)) * 64 + nt * 16];
            acc[nt] = wmma_f16(a, b, acc[nt]);
        }
    }
#pragma unroll
    for (int nt = 0; nt < 4; ++nt) {
        int n = n0 + nt * 16 + lo;
        float bv = bias[n];
#pragma unroll
        for (int r = 0; r < 8; ++r) {
            int m = m0 + r + 8 * hi;
            out[(size_t)m * 1024 + n] = acc[nt][r] + bv;
        }
    }
}

extern "C" void kernel_launch(void* const* d_in, const int* in_sizes, int n_in,
                              void* d_out, int out_size, void* d_ws, size_t ws_size,
                              hipStream_t stream) {
    const float* x    = (const float*)d_in[0];   // [4,2048,1024]
    const float* Wqkv = (const float*)d_in[1];   // [3072,1024]
    const float* bqkv = (const float*)d_in[2];   // [3072]
    const float* Wout = (const float*)d_in[3];   // [1024,1024]
    const float* bout = (const float*)d_in[4];   // [1024]
    float* out = (float*)d_out;

    char* ws = (char*)d_ws;
    _Float16* xh  = (_Float16*)(ws);                       // 16 MiB
    _Float16* wtq = (_Float16*)(ws + (16u << 20));         //  6 MiB
    _Float16* wto = (_Float16*)(ws + (22u << 20));         //  2 MiB
    _Float16* Qh  = (_Float16*)(ws + (24u << 20));         // 16 MiB
    _Float16* Kh  = (_Float16*)(ws + (40u << 20));         // 16 MiB
    _Float16* Vh  = (_Float16*)(ws + (56u << 20));         // 16 MiB
    _Float16* Oh  = (_Float16*)(ws + (72u << 20));         // 16 MiB  (total 88 MiB)

    cvt_f32_to_f16<<<4096, 256, 0, stream>>>(x, xh, (size_t)BB * SS * CC);
    transpose_f32_to_f16<<<dim3(1024 / 32, 3072 / 32), dim3(32, 8), 0, stream>>>(
        Wqkv, wtq, 3072, 1024);
    transpose_f32_to_f16<<<dim3(1024 / 32, 1024 / 32), dim3(32, 8), 0, stream>>>(
        Wout, wto, 1024, 1024);
    gemm_qkv<<<3072, 256, 0, stream>>>(xh, wtq, bqkv, Qh, Kh, Vh);
    attn_kernel<<<1024, 256, 0, stream>>>(Qh, Kh, Vh, Oh);
    gemm_out<<<1024, 256, 0, stream>>>(Oh, wto, bout, out);
}
